// cLSTM_27582279975488
// MI455X (gfx1250) — matrix operand order repaired and checked by reading, not compiled, not run on hardware
//
#include <hip/hip_runtime.h>
#include <math.h>

// ---------------------------------------------------------------------------
// Persistent 2-layer custom LSTM for MI455X (gfx1250, wave32, WMMA).
// B=64, S=512, D=256, H=512, gates 4H=2048.
//   layer0: gates = [x_t*exp(efw) | h0] @ W0 + b0   (K=768)
//   layer1: gates = [h0 | h1]           @ W1 + b1   (K=1024)
// bf16 WMMA with f32 accumulate; weights packed once into per-lane WMMA
// B-fragment layout; h double-buffered in global bf16; c-state in LDS;
// one software grid barrier per timestep; h panels staged into LDS with
// GLOBAL_LOAD_ASYNC_TO_LDS_B128 (ASYNCcnt) instead of VGPR round-trips.
// ---------------------------------------------------------------------------

typedef __attribute__((ext_vector_type(16))) __bf16 v16bf;
typedef __attribute__((ext_vector_type(8)))  __bf16 v8bf;
typedef __attribute__((ext_vector_type(8)))  float  v8f;

#define Bb   64
#define Ss   512
#define Dd   256
#define Hh   512
#define NGt  2048
#define NWG  32
#define TPB  256
#define ASTR 264   // padded LDS row stride in bf16 elements (16B aligned rows)

__device__ __forceinline__ float sigf(float z) { return 1.0f / (1.0f + __expf(-z)); }

// Monotonic-ticket grid barrier (all NWG blocks resident).
__device__ __forceinline__ void gbar(unsigned* bar) {
  __syncthreads();
  if (threadIdx.x == 0) {
    __threadfence();
    unsigned ticket = atomicAdd(bar, 1u);
    unsigned target = (ticket / NWG + 1u) * NWG;
    while (__hip_atomic_load(bar, __ATOMIC_RELAXED, __HIP_MEMORY_SCOPE_AGENT) < target) {
      __builtin_amdgcn_s_sleep(1);
    }
  }
  __syncthreads();
  __threadfence();  // acquire: invalidate near caches before reading peers' h
}

// Wave-relative LDS byte offset of a generic pointer to __shared__ data.
__device__ __forceinline__ unsigned lds_off_of(const void* p) {
  return (unsigned)(unsigned long long)(__attribute__((address_space(3))) const char*)p;
}

__device__ __forceinline__ void wait_asynccnt0() {
#if __has_builtin(__builtin_amdgcn_s_wait_asynccnt)
  __builtin_amdgcn_s_wait_asynccnt(0);
#else
  asm volatile("s_wait_asynccnt 0x0" ::: "memory");
#endif
}

// Stage a 64x256 bf16 slab of h (row stride Hh) into the LDS panel using the
// CDNA5 async L2->LDS DMA path (no VGPR data round-trip, tracked by ASYNCcnt).
__device__ __forceinline__ void stage_h_async(__bf16* As, const __bf16* __restrict__ src,
                                              int tid) {
  unsigned long long base = (unsigned long long)(uintptr_t)src;
#pragma unroll
  for (int it = 0; it < (Bb * 32) / TPB; ++it) {      // 8 x 16B chunks per thread
    int idx = tid + it * TPB;
    int m = idx >> 5;
    int c = (idx & 31) << 3;                          // 8 bf16 = 16B
    unsigned loff = lds_off_of(&As[m * ASTR + c]);
    unsigned voff = (unsigned)((m * Hh + c) * 2);     // byte offset from base
    asm volatile("global_load_async_to_lds_b128 %0, %1, %2"
                 :: "v"(loff), "v"(voff), "s"(base)
                 : "memory");
  }
  wait_asynccnt0();
}

__device__ __forceinline__ v16bf load_afrag(const __bf16* a) {
  v8bf lo = *(const v8bf*)a;
  v8bf hi = *(const v8bf*)(a + 16);
  return __builtin_shufflevector(lo, hi, 0,1,2,3,4,5,6,7,8,9,10,11,12,13,14,15);
}

// 8 K-chunks (K=256 panel) of bf16 WMMA for two 16x16 M-tiles of one N-tile.
// A-fragment LDS loads are software-pipelined one chunk ahead so the WMMAs
// overlap the ds_load latency of the next chunk.
__device__ __forceinline__ void do_kchunks(const __bf16* As, const __bf16* __restrict__ Wpk,
                                           int ktbase, int nt, int lane, int mh,
                                           v8f& acc0, v8f& acc1) {
  const int half = lane >> 4;
  const int mrow = lane & 15;
  const __bf16* abase = &As[(mh * 32 + mrow) * ASTR + half * 8];

  v16bf a0 = load_afrag(abase);
  v16bf a1 = load_afrag(abase + 16 * ASTR);
#pragma unroll
  for (int kc = 0; kc < 8; ++kc) {
    v16bf n0 = a0, n1 = a1;
    if (kc < 7) {                                     // prefetch next chunk's A
      n0 = load_afrag(abase + (kc + 1) * 32);
      n1 = load_afrag(abase + (kc + 1) * 32 + 16 * ASTR);
    }
    const long kt = (long)(ktbase + kc);
    v16bf bfrag = *(const v16bf*)(Wpk + ((kt * 128 + nt) * 32 + lane) * 16);
    acc0 = __builtin_amdgcn_wmma_f32_16x16x32_bf16(false, a0, false, bfrag,
                                                   (short)0, acc0, false, false);
    acc1 = __builtin_amdgcn_wmma_f32_16x16x32_bf16(false, a1, false, bfrag,
                                                   (short)0, acc1, false, false);
    a0 = n0;
    a1 = n1;
  }
}

// ---------------------------------------------------------------------------
// Setup kernels
// ---------------------------------------------------------------------------

__global__ void k_init(float* ef, const float* __restrict__ efw,
                       __bf16* h0, __bf16* h1, unsigned* bar) {
  int tid = blockIdx.x * blockDim.x + threadIdx.x;
  int nthr = gridDim.x * blockDim.x;
  for (int i = tid; i < Dd; i += nthr) ef[i] = __expf(efw[i]);
  for (int i = tid; i < 2 * Bb * Hh; i += nthr) { h0[i] = (__bf16)0.0f; h1[i] = (__bf16)0.0f; }
  for (int i = tid; i < 64; i += nthr) bar[i] = 0u;
}

// Pack W[K][2048] fp32 -> bf16 WMMA B-fragment layout:
// dst[(((kt*128)+nt)*32 + lane)*16 + e] = W[kt*32 + (lane>>4)*8 + (e&7) + (e>>3)*16][nt*16 + (lane&15)]
__global__ void k_pack(const float* __restrict__ W, __bf16* __restrict__ dst, int ktiles) {
  long total = (long)ktiles * 128 * 32 * 16;
  long stride = (long)gridDim.x * blockDim.x;
  for (long d = (long)blockIdx.x * blockDim.x + threadIdx.x; d < total; d += stride) {
    int  e  = (int)(d & 15);
    int  l  = (int)((d >> 4) & 31);
    int  nt = (int)((d >> 9) & 127);
    long kt = d >> 16;
    long k  = kt * 32 + ((l >> 4) << 3) + (e & 7) + ((e >> 3) << 4);
    int  n  = nt * 16 + (l & 15);
    dst[d] = (__bf16)W[k * NGt + n];
  }
}

// ---------------------------------------------------------------------------
// Persistent LSTM kernel: 32 blocks x 256 threads (8 wave32 per block)
// ---------------------------------------------------------------------------

__global__ __launch_bounds__(TPB, 1)
void k_lstm(const float* __restrict__ x, const float* __restrict__ b0,
            const float* __restrict__ b1, const __bf16* __restrict__ W0pk,
            const __bf16* __restrict__ W1pk, const float* __restrict__ ef,
            __bf16* __restrict__ h0base, __bf16* __restrict__ h1base,
            unsigned* __restrict__ bar, float* __restrict__ y) {
  __shared__ __bf16 As[Bb * ASTR];        // 33,792 B activation panel
  __shared__ float  gsm[4 * Bb * 16];     // 16,384 B gate exchange
  __shared__ float  c0s[Bb * 16];         //  4,096 B cell state L0
  __shared__ float  c1s[Bb * 16];         //  4,096 B cell state L1

  const int wg   = blockIdx.x;            // owns h-columns [wg*16, wg*16+16)
  const int tid  = threadIdx.x;
  const int wave = tid >> 5;
  const int lane = tid & 31;
  const int g    = wave & 3;              // gate: 0=i 1=f 2=g 3=o
  const int mh   = wave >> 2;             // M half (rows mh*32 .. mh*32+31)
  const int nt   = g * 32 + wg;           // global N-tile (of 128)
  const int half = lane >> 4;
  const int mrow = lane & 15;
  const float bias0 = b0[g * Hh + wg * 16 + mrow];
  const float bias1 = b1[g * Hh + wg * 16 + mrow];

  for (int i = tid; i < Bb * 16; i += TPB) { c0s[i] = 0.0f; c1s[i] = 0.0f; }
  __syncthreads();

  for (int s = 0; s < Ss; ++s) {
    const int pw = s & 1;        // write buffer
    const int pr = pw ^ 1;       // read buffer (previous step)
    const __bf16* h0r = h0base + pr * (Bb * Hh);
    __bf16*       h0w = h0base + pw * (Bb * Hh);
    const __bf16* h1r = h1base + pr * (Bb * Hh);
    __bf16*       h1w = h1base + pw * (Bb * Hh);

    // ======================= layer 0 (K = 768) =======================
    v8f acc0 = {0.f,0.f,0.f,0.f,0.f,0.f,0.f,0.f};
    v8f acc1 = {0.f,0.f,0.f,0.f,0.f,0.f,0.f,0.f};
    for (int p = 0; p < 3; ++p) {
      __syncthreads();
      if (p == 0) {
        // x_t slab with feature weighting: As[m][k] = bf16(x[m][s][k]*exp(efw[k]))
        for (int idx = tid; idx < Bb * (Dd / 4); idx += TPB) {
          int m = idx >> 6;
          int q = (idx & 63) << 2;
          const float4 xv = *(const float4*)(x + (size_t)m * Ss * Dd + (size_t)s * Dd + q);
          const float4 ev = *(const float4*)(ef + q);
          __bf16* dp = &As[m * ASTR + q];
          dp[0] = (__bf16)(xv.x * ev.x);
          dp[1] = (__bf16)(xv.y * ev.y);
          dp[2] = (__bf16)(xv.z * ev.z);
          dp[3] = (__bf16)(xv.w * ev.w);
        }
      } else {
        stage_h_async(As, h0r + (p - 1) * 256, tid);
      }
      __syncthreads();
      do_kchunks(As, W0pk, p * 8, nt, lane, mh, acc0, acc1);
    }
#pragma unroll
    for (int r = 0; r < 8; ++r) {
      gsm[(g * Bb + mh * 32 + half * 8 + r) * 16 + mrow]      = acc0[r] + bias0;
      gsm[(g * Bb + mh * 32 + 16 + half * 8 + r) * 16 + mrow] = acc1[r] + bias0;
    }
    __syncthreads();
    for (int idx = tid; idx < Bb * 16; idx += TPB) {
      float cprev = c0s[idx];
      float iv = sigf(gsm[idx]);
      float fv = sigf(gsm[Bb * 16 + idx] + cprev) * 0.9f;   // decayed forget gate
      float gv = tanhf(gsm[2 * Bb * 16 + idx]);
      float ov = sigf(gsm[3 * Bb * 16 + idx]);
      float c  = fv * cprev + iv * gv;
      float h  = ov * tanhf(c);
      c0s[idx] = c;
      int m = idx >> 4, j = idx & 15;
      h0w[m * Hh + wg * 16 + j] = (__bf16)h;
    }
    gbar(bar);  // h0(s) complete & visible device-wide

    // ======================= layer 1 (K = 1024) ======================
    acc0 = (v8f){0.f,0.f,0.f,0.f,0.f,0.f,0.f,0.f};
    acc1 = (v8f){0.f,0.f,0.f,0.f,0.f,0.f,0.f,0.f};
    for (int p = 0; p < 4; ++p) {
      __syncthreads();
      const __bf16* src = (p < 2) ? (h0w + p * 256) : (h1r + (p - 2) * 256);
      stage_h_async(As, src, tid);
      __syncthreads();
      do_kchunks(As, W1pk, p * 8, nt, lane, mh, acc0, acc1);
    }
#pragma unroll
    for (int r = 0; r < 8; ++r) {
      gsm[(g * Bb + mh * 32 + half * 8 + r) * 16 + mrow]      = acc0[r] + bias1;
      gsm[(g * Bb + mh * 32 + 16 + half * 8 + r) * 16 + mrow] = acc1[r] + bias1;
    }
    __syncthreads();
    for (int idx = tid; idx < Bb * 16; idx += TPB) {
      float cprev = c1s[idx];
      float iv = sigf(gsm[idx]);
      float fv = sigf(gsm[Bb * 16 + idx] + cprev) * 0.9f;
      float gv = tanhf(gsm[2 * Bb * 16 + idx]);
      float ov = sigf(gsm[3 * Bb * 16 + idx]);
      float c  = fv * cprev + iv * gv;
      float h  = ov * tanhf(c);
      c1s[idx] = c;
      int m = idx >> 4, j = idx & 15;
      h1w[m * Hh + wg * 16 + j] = (__bf16)h;
      y[(size_t)m * Ss * Hh + (size_t)s * Hh + wg * 16 + j] = h;  // [B,S,H] f32
    }
    // No trailing barrier needed: next step's mid-step barrier orders
    // h1(s) writes before any WG's layer-1(s+1) reads (double buffering).
  }
}

// ---------------------------------------------------------------------------
// Launch
// ---------------------------------------------------------------------------

extern "C" void kernel_launch(void* const* d_in, const int* in_sizes, int n_in,
                              void* d_out, int out_size, void* d_ws, size_t ws_size,
                              hipStream_t stream) {
  (void)in_sizes; (void)n_in; (void)out_size; (void)ws_size;
  const float* x   = (const float*)d_in[0];
  const float* W0  = (const float*)d_in[1];
  const float* b0  = (const float*)d_in[2];
  const float* W1  = (const float*)d_in[3];
  const float* b1  = (const float*)d_in[4];
  const float* efw = (const float*)d_in[5];

  char* ws = (char*)d_ws;
  __bf16*   W0pk = (__bf16*)(ws + 0);          // 768*2048*2  = 3,145,728 B
  __bf16*   W1pk = (__bf16*)(ws + 3145728);    // 1024*2048*2 = 4,194,304 B
  float*    ef   = (float*)(ws + 7340032);     // 256*4       = 1,024 B
  __bf16*   h0   = (__bf16*)(ws + 7341056);    // 2*64*512*2  = 131,072 B
  __bf16*   h1   = (__bf16*)(ws + 7472128);    // 2*64*512*2  = 131,072 B
  unsigned* bar  = (unsigned*)(ws + 7603200);  // 256 B        (~7.6 MB total)
  float*    y    = (float*)d_out;

  k_init<<<dim3(64), dim3(256), 0, stream>>>(ef, efw, h0, h1, bar);
  k_pack<<<dim3(1024), dim3(256), 0, stream>>>(W0, W0pk, 24);  // K=768  -> 24 ktiles
  k_pack<<<dim3(1024), dim3(256), 0, stream>>>(W1, W1pk, 32);  // K=1024 -> 32 ktiles
  k_lstm<<<dim3(NWG), dim3(TPB), 0, stream>>>(x, b0, b1, W0pk, W1pk, ef, h0, h1, bar, y);
}
